// EnhancedGapLoss_33234456937173
// MI455X (gfx1250) — compile-verified, matrix operand order checked by reading, not run on hardware
//
#include <hip/hip_runtime.h>
#include <hip/hip_bf16.h>

typedef __attribute__((ext_vector_type(16))) _Float16 v16h;
typedef __attribute__((ext_vector_type(8)))  float    v8f;

#if __has_builtin(__builtin_amdgcn_global_load_async_to_lds_b128) && \
    __has_builtin(__builtin_amdgcn_s_wait_asynccnt)
#define USE_ASYNC_LDS 1
typedef __attribute__((ext_vector_type(4))) int v4i;
typedef __attribute__((address_space(1))) v4i* gbl_v4ip;
typedef __attribute__((address_space(3))) v4i* lds_v4ip;
#endif

#define B_   4
#define H_   512
#define W_   512
#define HW_  (H_ * W_)                 // 262144
#define NPIX (B_ * HW_)                // 1048576
#define K_PARAM 20.0f
#define CONT_W  0.3f
#define DIR_W   0.5f
#define INF1    1024.0f                // H + W
#define D0INIT  2097152.0f             // 2*(H+W)^2

// ---------------- workspace layout (bytes) ----------------
#define OFF_L    (0)
#define OFF_G2   (4u << 20)
#define OFF_DIST (8u << 20)
#define OFF_A    (12u << 20)
#define OFF_SKEL (13u << 20)
#define OFF_ACC  (14u << 20)

// ---------------------------------------------------------------------------
__global__ void gl_init(float* acc) {
    if (threadIdx.x == 0) { acc[0] = 0.0f; acc[1] = 0.0f; acc[2] = 0.0f; acc[3] = 0.0f; }
}

// ---------------------------------------------------------------------------
// Fused 2-class log-softmax cross-entropy + argmax.
__global__ void gl_ce_argmax(const float* __restrict__ pred,
                             const int* __restrict__ target,
                             float* __restrict__ L,
                             unsigned char* __restrict__ A) {
    for (int i = blockIdx.x * blockDim.x + threadIdx.x; i < NPIX;
         i += gridDim.x * blockDim.x) {
        int b = i >> 18;
        int hw = i & (HW_ - 1);
        const float* p0 = pred + ((size_t)b * 2) * HW_;
        __builtin_prefetch(p0 + hw + 4096, 0, 0);   // global_prefetch_b8
        float x0 = p0[hw];
        float x1 = p0[HW_ + hw];
        int   t  = target[i];
        float m  = fmaxf(x0, x1);
        float lse = m + logf(expf(x0 - m) + expf(x1 - m));
        L[i] = lse - (t ? x1 : x0);
        A[i] = (x1 > x0) ? 1 : 0;
    }
}

// ---------------------------------------------------------------------------
// Zhang-Suen thinning, one image per workgroup, fully LDS-resident.
// Padded image: 514 rows x 544B stride, interior at byte 16 of each row
// (zero border -> branch-free neighbors; 16B alignment -> async B128 loads).
// 273KB image + 32KB removal mask fits CDNA5's 320KB WGP LDS.
#define SPW    544
#define SIMG   (514 * SPW)              // 279616
#define SOFF(h, w) (((h) + 1) * SPW + 16 + (w))
#define NWORDS (HW_ / 32)               // 8192

__device__ __forceinline__ int zs_remove_p(const unsigned char* img, int p, int first) {
    const int h = p >> 9, w = p & (W_ - 1);
    const int base = SOFF(h, w);
    if (!img[base]) return 0;
    const int P2 = img[base - SPW],     P3 = img[base - SPW + 1];
    const int P4 = img[base + 1],       P5 = img[base + SPW + 1];
    const int P6 = img[base + SPW],     P7 = img[base + SPW - 1];
    const int P8 = img[base - 1],       P9 = img[base - SPW - 1];
    const int bsum = P2 + P3 + P4 + P5 + P6 + P7 + P8 + P9;
    if (bsum < 2 || bsum > 6) return 0;
    const int a = ((P2 == 0) & (P3 == 1)) + ((P3 == 0) & (P4 == 1)) +
                  ((P4 == 0) & (P5 == 1)) + ((P5 == 0) & (P6 == 1)) +
                  ((P6 == 0) & (P7 == 1)) + ((P7 == 0) & (P8 == 1)) +
                  ((P8 == 0) & (P9 == 1)) + ((P9 == 0) & (P2 == 1));
    if (a != 1) return 0;
    return first ? (((P2 & P4 & P6) == 0) & ((P4 & P6 & P8) == 0))
                 : (((P2 & P4 & P8) == 0) & ((P2 & P6 & P8) == 0));
}

__global__ void gl_skeletonize(const unsigned char* __restrict__ Ain,
                               unsigned char* __restrict__ skelOut) {
    extern __shared__ unsigned char lds[];
    unsigned char* img = lds;                                   // SIMG bytes
    unsigned int*  rm  = (unsigned int*)(lds + SIMG);           // 32768 B
    int* changed       = (int*)(lds + SIMG + NWORDS * 4);       // 4 B

    const int tid = threadIdx.x;
    const int nt  = blockDim.x;
    const unsigned char* src = Ain + (size_t)blockIdx.x * HW_;
    unsigned char* dst       = skelOut + (size_t)blockIdx.x * HW_;

    for (int p = tid; p < SIMG; p += nt) img[p] = 0;            // zero border
    __syncthreads();
#ifdef USE_ASYNC_LDS
    // async DMA of the 256KB image straight into LDS (no VGPR round-trip)
    for (int ck = tid; ck < H_ * 32; ck += nt) {                // 16B chunks
        const int h = ck >> 5, c16 = (ck & 31) << 4;
        __builtin_amdgcn_global_load_async_to_lds_b128(
            (gbl_v4ip)(src + (h << 9) + c16),
            (lds_v4ip)(img + SOFF(h, 0) + c16), 0, 0);
    }
    __builtin_amdgcn_s_wait_asynccnt(0);
#else
    for (int p = tid; p < HW_; p += nt)
        img[SOFF(p >> 9, p & (W_ - 1))] = src[p];
#endif
    __syncthreads();

    for (int iter = 0; iter < 1024; ++iter) {
        if (tid == 0) *changed = 0;
        __syncthreads();
#pragma unroll
        for (int sub = 0; sub < 2; ++sub) {
            const int first = (sub == 0);
            unsigned any = 0;
            for (int word = tid; word < NWORDS; word += nt) {   // thread owns word
                const int pb = word << 5;
                unsigned m = 0;
#pragma unroll 4
                for (int bit = 0; bit < 32; ++bit)
                    m |= (unsigned)zs_remove_p(img, pb + bit, first) << bit;
                rm[word] = m;
                any |= m;
            }
            __syncthreads();
            for (int word = tid; word < NWORDS; word += nt) {   // apply simultaneously
                unsigned m = rm[word];
                const int pb = word << 5;
                while (m) {
                    const int bit = __builtin_ctz(m);
                    const int p = pb + bit;
                    img[SOFF(p >> 9, p & (W_ - 1))] = 0;
                    m &= m - 1;
                }
            }
            if (any) atomicOr(changed, 1);
            __syncthreads();
        }
        if (*changed == 0) break;
        __syncthreads();
    }

    for (int p = tid; p < HW_; p += nt)
        dst[p] = img[SOFF(p >> 9, p & (W_ - 1))];
}

// ---------------------------------------------------------------------------
// EDT pass 1: vertical forward/backward scan per column; writes g^2.
__global__ void gl_edt_vert(const unsigned char* __restrict__ skel,
                            float* __restrict__ g2) {
    int col = blockIdx.x * blockDim.x + threadIdx.x;
    if (col >= B_ * W_) return;
    int b = col >> 9, w = col & (W_ - 1);
    const unsigned char* sk = skel + ((size_t)b << 18);
    float* g = g2 + ((size_t)b << 18);
    float d = INF1;
    for (int h = 0; h < H_; ++h) {
        int idx = (h << 9) | w;
        d = sk[idx] ? 0.0f : fminf(d + 1.0f, INF1);
        g[idx] = d;
    }
    d = INF1;
    for (int h = H_ - 1; h >= 0; --h) {
        int idx = (h << 9) | w;
        d = sk[idx] ? 0.0f : fminf(d + 1.0f, INF1);
        float gm = fminf(g[idx], d);
        g[idx] = gm * gm;
    }
}

// EDT pass 2: per-row relax over columns (row cached in LDS), dist = sqrt.
__global__ void gl_edt_horiz(const float* __restrict__ g2,
                             float* __restrict__ dist) {
    __shared__ float srow[W_];
    const int row = blockIdx.x;
    const int j   = threadIdx.x;
    const size_t base = (size_t)row * W_;
    srow[j] = g2[base + j];
    __syncthreads();
    float best = D0INIT;
    const float jf = (float)j;
#pragma unroll 4
    for (int q = 0; q < W_; ++q) {
        float dq = jf - (float)q;
        best = fminf(best, fmaf(dq, dq, srow[q]));
    }
    dist[base + j] = sqrtf(best);
}

// ---------------------------------------------------------------------------
// Fused endpoint / Wmap / continuity / direction terms + WMMA wave reduction.
// Block = 8-row strip; skeleton strip + halo async-DMA'd into LDS (row stride
// 544, interior at byte 16 -> aligned B128). Each wave owns one 512-px row ->
// one v_wmma_f32_16x16x32_f16 with A=ones sums all 512 contributions.
#define TROWS 8
#define FPW   544
#define FBYTES ((TROWS + 2) * FPW)              // 5440

__global__ void __launch_bounds__(256)
gl_fuse_reduce(const unsigned char* __restrict__ skel,
               const float* __restrict__ dist,
               const float* __restrict__ L,
               float* __restrict__ acc) {
    __shared__ __attribute__((aligned(16))) unsigned char tile[FBYTES];

    const int T  = blockIdx.x;                  // 0..255 (64 strips/image)
    const int b  = T >> 6;
    const int r0 = (T & 63) * TROWS;
    const unsigned char* sk = skel + ((size_t)b << 18);

    for (int idx = threadIdx.x; idx < FBYTES; idx += blockDim.x) tile[idx] = 0;
    __syncthreads();
#ifdef USE_ASYNC_LDS
    for (int ck = threadIdx.x; ck < (TROWS + 2) * 32; ck += blockDim.x) {
        const int lr = ck >> 5, c16 = (ck & 31) << 4;
        const int gr = r0 + lr - 1;
        if ((unsigned)gr < (unsigned)H_)
            __builtin_amdgcn_global_load_async_to_lds_b128(
                (gbl_v4ip)(sk + (gr << 9) + c16),
                (lds_v4ip)(tile + lr * FPW + 16 + c16), 0, 0);
    }
    __builtin_amdgcn_s_wait_asynccnt(0);
#else
    for (int idx = threadIdx.x; idx < (TROWS + 2) * W_; idx += blockDim.x) {
        const int lr = idx >> 9, lc = idx & (W_ - 1);
        const int gr = r0 + lr - 1;
        if ((unsigned)gr < (unsigned)H_)
            tile[lr * FPW + 16 + lc] = sk[(gr << 9) | lc];
    }
#endif
    __syncthreads();

    const int lane = threadIdx.x & 31;
    const int wv   = threadIdx.x >> 5;          // wave -> row within strip
    const int grow = r0 + wv;
    const size_t rowBase = (((size_t)b << 18) | ((size_t)grow << 9)) + lane * 16;

    float Lb_[16], Db_[16];
    {
        const float4* L4 = (const float4*)(L + rowBase);
        const float4* D4 = (const float4*)(dist + rowBase);
#pragma unroll
        for (int k = 0; k < 4; ++k) {           // b128 coalesced loads
            *(float4*)&Lb_[k * 4] = L4[k];
            *(float4*)&Db_[k * 4] = D4[k];
        }
    }

    const unsigned char* rU = tile + (wv + 0) * FPW + 16 + lane * 16;  // above
    const unsigned char* rC = tile + (wv + 1) * FPW + 16 + lane * 16;  // center
    const unsigned char* rD = tile + (wv + 2) * FPW + 16 + lane * 16;  // below

    v16h bv;
#pragma unroll
    for (int e = 0; e < 16; ++e) {
        const int cc = rC[e];
        const int up = rU[e],     dn = rD[e];
        const int lf = rC[e - 1], rt = rC[e + 1];
        const int nw = rU[e - 1], ne = rU[e + 1];
        const int sw = rD[e - 1], se = rD[e + 1];
        const int ring = up + dn + lf + rt + nw + ne + sw + se;
        const int Cv   = cc * ring;
        const float endp = (Cv == 1 || Cv >= 3) ? 1.0f : 0.0f;
        const float Wmap = expf(-Db_[e] * (1.0f / K_PARAM)) + endp * K_PARAM;
        const int r1 = lf + cc + rt;            // 1x3
        const int r2 = up + cc + dn;            // 3x1
        const int r3 = nw + cc + se;            // main diag
        const int r4 = ne + cc + sw;            // anti diag
        const int cont = abs(r1 - cc) + abs(r2 - cc) + abs(r3 - cc) + abs(r4 - cc);
        const int dirl = abs(1 - r1) + abs(1 - r2) + abs(1 - r3) + abs(1 - r4);
        const float v = Wmap * Lb_[e] + CONT_W * (float)cont + DIR_W * (float)dirl;
        bv[e] = (_Float16)v;
    }

    v16h ones;
#pragma unroll
    for (int i = 0; i < 16; ++i) ones[i] = (_Float16)1.0f;
    v8f c = {};
    // D = ones(16x32) * B + 0 : every row of D = column sums; sum(D) = 16*total
    c = __builtin_amdgcn_wmma_f32_16x16x32_f16(
            false, ones, false, bv, (short)0, c, false, false);

    float s = c[0] + c[1] + c[2] + c[3] + c[4] + c[5] + c[6] + c[7];
#pragma unroll
    for (int off = 16; off > 0; off >>= 1) s += __shfl_xor(s, off, 32);
    if (lane == 0) atomicAdd(acc, s * (1.0f / 16.0f));   // global_atomic_add_f32
}

// ---------------------------------------------------------------------------
__global__ void gl_finalize(const float* __restrict__ acc, float* __restrict__ out) {
    if (threadIdx.x == 0) out[0] = acc[0] * (1.0f / (float)NPIX);
}

// ---------------------------------------------------------------------------
extern "C" void kernel_launch(void* const* d_in, const int* in_sizes, int n_in,
                              void* d_out, int out_size, void* d_ws, size_t ws_size,
                              hipStream_t stream) {
    const float* pred   = (const float*)d_in[0];   // (4,2,512,512) f32
    const int*   target = (const int*)d_in[1];     // (4,512,512) i32
    float* out = (float*)d_out;

    unsigned char* ws = (unsigned char*)d_ws;
    float*         Lb   = (float*)(ws + OFF_L);
    float*         g2   = (float*)(ws + OFF_G2);
    float*         dist = (float*)(ws + OFF_DIST);
    unsigned char* Ab   = ws + OFF_A;
    unsigned char* sk   = ws + OFF_SKEL;
    float*         acc  = (float*)(ws + OFF_ACC);

    gl_init<<<1, 32, 0, stream>>>(acc);

    gl_ce_argmax<<<2048, 256, 0, stream>>>(pred, target, Lb, Ab);

    // one image per WGP: padded image + removal mask resident in 320KB LDS
    const size_t skLds = SIMG + NWORDS * 4 + 16;    // ~312 KB
    gl_skeletonize<<<B_, 1024, skLds, stream>>>(Ab, sk);

    gl_edt_vert<<<(B_ * W_ + 255) / 256, 256, 0, stream>>>(sk, g2);
    gl_edt_horiz<<<B_ * H_, W_, 0, stream>>>(g2, dist);

    gl_fuse_reduce<<<256, 256, 0, stream>>>(sk, dist, Lb, acc);

    gl_finalize<<<1, 32, 0, stream>>>(acc, out);
}
